// Decoder_fusion_22960895165097
// MI455X (gfx1250) — matrix operand order repaired and checked by reading, not compiled, not run on hardware
//
#include <hip/hip_runtime.h>
#include <math.h>

#define BB 8
#define CC 128
#define LL 4096
#define CK 512

typedef __bf16 bf16_t;
typedef __attribute__((ext_vector_type(16))) __bf16 v16bf;
typedef __attribute__((ext_vector_type(8)))  float  v8f;

__device__ __forceinline__ bf16_t f2bf(float x) {
    unsigned u = __builtin_bit_cast(unsigned, x);
    u = (u + 0x7FFFu + ((u >> 16) & 1u)) >> 16;      // round-to-nearest-even
    unsigned short s = (unsigned short)u;
    return __builtin_bit_cast(bf16_t, s);
}

__device__ __forceinline__ float bf2f(bf16_t b) {
    unsigned u = ((unsigned)__builtin_bit_cast(unsigned short, b)) << 16;
    return __builtin_bit_cast(float, u);
}

__device__ __forceinline__ float gelu_exact(float y) {
    return 0.5f * y * (1.f + erff(y * 0.70710678118654752f));
}

// CDNA5 async global->LDS copy (16 bytes / lane), tracked by ASYNCcnt.
// LDS byte address = low 32 bits of the generic shared pointer (ISA 10.2).
__device__ __forceinline__ void async_b128(const void* gptr, void* lptr) {
    unsigned lds = (unsigned)(unsigned long long)lptr;
    unsigned long long g = (unsigned long long)gptr;
    asm volatile("global_load_async_to_lds_b128 %0, %1, off"
                 :: "v"(lds), "v"(g) : "memory");
}
__device__ __forceinline__ void async_wait0() {
    asm volatile("s_wait_asynccnt 0" ::: "memory");
}

__device__ __forceinline__ void store_out(float* p, float v)  { *p = v; }
__device__ __forceinline__ void store_out(bf16_t* p, float v) { *p = f2bf(v); }

// ---------------------------------------------------------------------------
// Pointwise (1x1x1) conv as WMMA GEMM + bias + LayerNorm(+GELU)(+residual).
// Workgroup: 8 waves; tile = M(=Cout)128 x N(=voxels)64, K(=Cin)=128.
// TI=float: stage X with VALU convert.  TI=bf16: async global->LDS copy.
// ---------------------------------------------------------------------------
template<typename TI, typename TO, bool GELU, bool RESID>
__global__ __launch_bounds__(256) void pw_block_kernel(
    const TI* __restrict__ X,        // [B, 128, L]
    const float* __restrict__ W,     // [128, 128] (o, c)
    const float* __restrict__ bias,
    const float* __restrict__ gam,
    const float* __restrict__ bet,
    const float* __restrict__ resid, // [B, 128, L] fp32 or null
    TO* __restrict__ Y,              // [B, Ctot, L]
    int Ctot, int coff)
{
    __shared__ bf16_t Wl[CC][CC];    // 32 KB
    __shared__ bf16_t Xl[CC][64];    // 16 KB
    __shared__ float s_sum[64], s_ssq[64], s_mu[64], s_rs[64];

    const int tid = threadIdx.x;
    const int b  = blockIdx.y;
    const int l0 = blockIdx.x * 64;

    const TI* Xb = X + (size_t)b * CC * LL + l0;
    if constexpr (__is_same(TI, float)) {
        #pragma unroll 8
        for (int i = 0; i < 32; ++i) {
            int e = tid + i * 256;
            Xl[e >> 6][e & 63] = f2bf(Xb[(size_t)(e >> 6) * LL + (e & 63)]);
        }
    } else {
        #pragma unroll
        for (int i = 0; i < 4; ++i) {               // 1024 x 16B transfers
            int tr = tid + i * 256;
            int row = tr >> 3, seg = tr & 7;
            async_b128(Xb + (size_t)row * LL + seg * 8, &Xl[row][seg * 8]);
        }
    }
    #pragma unroll 8
    for (int i = 0; i < 64; ++i) {
        int e = tid + i * 256;
        Wl[e >> 7][e & 127] = f2bf(W[e]);
    }
    if (tid < 64) { s_sum[tid] = 0.f; s_ssq[tid] = 0.f; }
    if constexpr (!__is_same(TI, float)) async_wait0();
    __syncthreads();

    const int wave = tid >> 5, lane = tid & 31;
    const int lo = lane & 15, hi = lane >> 4;
    const int mb = wave * 16;

    v8f acc[4];
    #pragma unroll
    for (int t = 0; t < 4; ++t)
        #pragma unroll
        for (int e = 0; e < 8; ++e) acc[t][e] = 0.f;

    #pragma unroll
    for (int kk = 0; kk < CC; kk += 32) {
        v16bf a;                                  // A: 16-bit 16x32 layout
        #pragma unroll
        for (int e = 0; e < 16; ++e)
            a[e] = Wl[mb + lo][kk + e + (hi << 4)];
        #pragma unroll
        for (int nt = 0; nt < 4; ++nt) {
            v16bf bm;                             // B: 32x16, row=k, col=n
            #pragma unroll
            for (int e = 0; e < 16; ++e)
                bm[e] = Xl[kk + e + (hi << 4)][nt * 16 + lo];
            acc[nt] = __builtin_amdgcn_wmma_f32_16x16x32_bf16(
                false, a, false, bm, (short)0, acc[nt], false, false);
        }
    }

    // bias + cross-wave LN statistics (channel dim spans all 8 waves)
    #pragma unroll
    for (int nt = 0; nt < 4; ++nt) {
        float ps = 0.f, pq = 0.f;
        #pragma unroll
        for (int e = 0; e < 8; ++e) {
            int m = mb + e + hi * 8;
            float v = acc[nt][e] + bias[m];
            acc[nt][e] = v;
            ps += v; pq += v * v;
        }
        int n = nt * 16 + lo;
        atomicAdd(&s_sum[n], ps);
        atomicAdd(&s_ssq[n], pq);
    }
    __syncthreads();
    if (tid < 64) {
        float mu  = s_sum[tid] * (1.f / CC);
        float var = s_ssq[tid] * (1.f / CC) - mu * mu;
        s_mu[tid] = mu;
        s_rs[tid] = rsqrtf(var + 1e-6f);
    }
    __syncthreads();

    TO* Yb = Y + ((size_t)b * Ctot + coff) * LL + l0;
    #pragma unroll
    for (int nt = 0; nt < 4; ++nt) {
        int n = nt * 16 + lo;
        float mu = s_mu[n], rs = s_rs[n];
        #pragma unroll
        for (int e = 0; e < 8; ++e) {
            int m = mb + e + hi * 8;
            float y = (acc[nt][e] - mu) * rs * gam[m] + bet[m];
            if constexpr (GELU) y = gelu_exact(y);
            if constexpr (RESID) y += resid[((size_t)b * CC + m) * LL + l0 + n];
            store_out(&Yb[(size_t)m * LL + n], y);
        }
    }
}

// ---------------------------------------------------------------------------
// Depthwise 3x3x3 conv + bias + LN + GELU. One voxel per 128-thread block.
// bf16 in / bf16 out, fp32 math.
// ---------------------------------------------------------------------------
__global__ __launch_bounds__(128) void dw_block_kernel(
    const bf16_t* __restrict__ X, const float* __restrict__ Wd,  // [128, 27]
    const float* __restrict__ bias, const float* __restrict__ gam,
    const float* __restrict__ bet, bf16_t* __restrict__ Y)
{
    const int v = blockIdx.x;
    const int b = blockIdx.y;
    const int c = threadIdx.x;
    const int d = v >> 8, h = (v >> 4) & 15, w = v & 15;

    const bf16_t* Xc = X + ((size_t)b * CC + c) * LL;
    const float* wp = Wd + c * 27;
    float sum = 0.f;
    #pragma unroll
    for (int kd = 0; kd < 3; ++kd) {
        int dd = d + kd - 1;
        if ((unsigned)dd >= 16u) continue;
        #pragma unroll
        for (int kh = 0; kh < 3; ++kh) {
            int hh = h + kh - 1;
            if ((unsigned)hh >= 16u) continue;
            #pragma unroll
            for (int kw = 0; kw < 3; ++kw) {
                int ww = w + kw - 1;
                if ((unsigned)ww >= 16u) continue;
                sum += wp[kd * 9 + kh * 3 + kw] * bf2f(Xc[dd * 256 + hh * 16 + ww]);
            }
        }
    }
    float val = sum + bias[c];

    float s = val, q = val * val;
    #pragma unroll
    for (int off = 16; off > 0; off >>= 1) {
        s += __shfl_xor(s, off, 32);
        q += __shfl_xor(q, off, 32);
    }
    __shared__ float ws_[4], wq_[4], s_mu, s_rs;
    const int wave = threadIdx.x >> 5;
    if ((threadIdx.x & 31) == 0) { ws_[wave] = s; wq_[wave] = q; }
    __syncthreads();
    if (threadIdx.x == 0) {
        float S = ws_[0] + ws_[1] + ws_[2] + ws_[3];
        float Q = wq_[0] + wq_[1] + wq_[2] + wq_[3];
        float mu = S * (1.f / CC);
        s_mu = mu;
        s_rs = rsqrtf(Q * (1.f / CC) - mu * mu + 1e-6f);
    }
    __syncthreads();
    float y = (val - s_mu) * s_rs * gam[c] + bet[c];
    Y[((size_t)b * CC + c) * LL + v] = f2bf(gelu_exact(y));
}

// ---------------------------------------------------------------------------
// attn = (q . K^T) / sqrt(L).  M=128 query ch, N=128-tile of 512 key ch,
// K-loop over L=4096 in chunks of 32.  bf16 inputs staged by async copy.
// ---------------------------------------------------------------------------
__global__ __launch_bounds__(256) void qk_kernel(
    const bf16_t* __restrict__ Q, const bf16_t* __restrict__ Kc,
    float* __restrict__ attn)
{
    __shared__ bf16_t ql[CC][32];   // 8 KB
    __shared__ bf16_t kl[CC][32];   // 8 KB
    const int tid = threadIdx.x;
    const int kt = blockIdx.x;
    const int b  = blockIdx.y;
    const int wave = tid >> 5, lane = tid & 31;
    const int lo = lane & 15, hi = lane >> 4;
    const int mb = wave * 16;

    v8f acc[8];
    #pragma unroll
    for (int t = 0; t < 8; ++t)
        #pragma unroll
        for (int e = 0; e < 8; ++e) acc[t][e] = 0.f;

    const bf16_t* Qb = Q  + (size_t)b * CC * LL;
    const bf16_t* Kb = Kc + (size_t)b * CK * LL + (size_t)kt * 128 * LL;

    for (int l0 = 0; l0 < LL; l0 += 32) {
        #pragma unroll
        for (int i = 0; i < 2; ++i) {           // 512 x 16B per matrix
            int tr = tid + i * 256;
            int row = tr >> 2, seg = tr & 3;
            async_b128(Qb + (size_t)row * LL + l0 + seg * 8, &ql[row][seg * 8]);
            async_b128(Kb + (size_t)row * LL + l0 + seg * 8, &kl[row][seg * 8]);
        }
        async_wait0();
        __syncthreads();
        v16bf a;
        #pragma unroll
        for (int e = 0; e < 16; ++e)
            a[e] = ql[mb + lo][e + (hi << 4)];
        #pragma unroll
        for (int nt = 0; nt < 8; ++nt) {
            v16bf bm;
            #pragma unroll
            for (int e = 0; e < 16; ++e)
                bm[e] = kl[nt * 16 + lo][e + (hi << 4)];
            acc[nt] = __builtin_amdgcn_wmma_f32_16x16x32_bf16(
                false, a, false, bm, (short)0, acc[nt], false, false);
        }
        __syncthreads();
    }

    float* Ab = attn + (size_t)b * CC * CK + kt * 128;
    #pragma unroll
    for (int nt = 0; nt < 8; ++nt)
        #pragma unroll
        for (int e = 0; e < 8; ++e) {
            int m = mb + e + hi * 8;
            Ab[(size_t)m * CK + nt * 16 + lo] = acc[nt][e] * 0.015625f; // L^-0.5
        }
}

// ---------------------------------------------------------------------------
// Masked softmax over the 512 key-channels; one wave per (b, c) row.
// ---------------------------------------------------------------------------
__global__ __launch_bounds__(128) void softmax_kernel(
    float* __restrict__ attn, const int* __restrict__ mask)
{
    const int wave = threadIdx.x >> 5, lane = threadIdx.x & 31;
    const int row = blockIdx.x * 4 + wave;
    const int b = row >> 7;
    float* p = attn + (size_t)row * CK;

    float v[16];
    float mx = -__builtin_inff();
    #pragma unroll
    for (int j = 0; j < 16; ++j) {
        int k = lane + j * 32;
        float x = p[k];
        if (mask[b * 4 + (k >> 7)] == 0) x = -__builtin_inff();
        v[j] = x;
        mx = fmaxf(mx, x);
    }
    #pragma unroll
    for (int off = 16; off > 0; off >>= 1) mx = fmaxf(mx, __shfl_xor(mx, off, 32));
    float sum = 0.f;
    #pragma unroll
    for (int j = 0; j < 16; ++j) { v[j] = expf(v[j] - mx); sum += v[j]; }
    #pragma unroll
    for (int off = 16; off > 0; off >>= 1) sum += __shfl_xor(sum, off, 32);
    float inv = 1.f / sum;
    #pragma unroll
    for (int j = 0; j < 16; ++j) p[lane + j * 32] = v[j] * inv;
}

// ---------------------------------------------------------------------------
// x = attn . V : M=128, N=128-tile of L, K-loop over 512 in chunks of 32.
// attn fp32 (VALU convert), V bf16 (async copy), bf16 out.
// ---------------------------------------------------------------------------
__global__ __launch_bounds__(256) void av_kernel(
    const float* __restrict__ attn, const bf16_t* __restrict__ V,
    bf16_t* __restrict__ Xo)
{
    __shared__ bf16_t al[CC][32];    // 8 KB
    __shared__ bf16_t vl[32][128];   // 8 KB
    const int tid = threadIdx.x;
    const int nt0 = blockIdx.x;      // l-tile of 128
    const int b = blockIdx.y;
    const int wave = tid >> 5, lane = tid & 31;
    const int lo = lane & 15, hi = lane >> 4;
    const int mb = wave * 16;

    v8f acc[8];
    #pragma unroll
    for (int t = 0; t < 8; ++t)
        #pragma unroll
        for (int e = 0; e < 8; ++e) acc[t][e] = 0.f;

    const float*  Ab = attn + (size_t)b * CC * CK;
    const bf16_t* Vb = V + (size_t)b * CK * LL + (size_t)nt0 * 128;

    for (int k0 = 0; k0 < CK; k0 += 32) {
        #pragma unroll
        for (int i = 0; i < 2; ++i) {           // V: 512 x 16B transfers
            int tr = tid + i * 256;
            int row = tr >> 4, seg = tr & 15;
            async_b128(Vb + (size_t)(k0 + row) * LL + seg * 8, &vl[row][seg * 8]);
        }
        #pragma unroll
        for (int i = 0; i < 16; ++i) {          // attn chunk fp32 -> bf16
            int e = tid + i * 256;
            al[e >> 5][e & 31] = f2bf(Ab[(size_t)(e >> 5) * CK + k0 + (e & 31)]);
        }
        async_wait0();
        __syncthreads();
        v16bf a;
        #pragma unroll
        for (int e = 0; e < 16; ++e)
            a[e] = al[mb + lo][e + (hi << 4)];
        #pragma unroll
        for (int nt = 0; nt < 8; ++nt) {
            v16bf bm;
            #pragma unroll
            for (int e = 0; e < 16; ++e)
                bm[e] = vl[e + (hi << 4)][nt * 16 + lo];
            acc[nt] = __builtin_amdgcn_wmma_f32_16x16x32_bf16(
                false, a, false, bm, (short)0, acc[nt], false, false);
        }
        __syncthreads();
    }

    #pragma unroll
    for (int nt = 0; nt < 8; ++nt)
        #pragma unroll
        for (int e = 0; e < 8; ++e) {
            int m = mb + e + hi * 8;
            Xo[((size_t)b * CC + m) * LL + nt0 * 128 + nt * 16 + lo] = f2bf(acc[nt][e]);
        }
}

// ---------------------------------------------------------------------------
extern "C" void kernel_launch(void* const* d_in, const int* in_sizes, int n_in,
                              void* d_out, int out_size, void* d_ws, size_t ws_size,
                              hipStream_t stream)
{
    const float* query = (const float*)d_in[0];
    const float* mods[4] = {(const float*)d_in[1], (const float*)d_in[2],
                            (const float*)d_in[3], (const float*)d_in[4]};
    const int*   mask  = (const int*)d_in[5];
    const float* pw1_w = (const float*)d_in[6];
    const float* pw1_b = (const float*)d_in[7];
    const float* ln1_g = (const float*)d_in[8];
    const float* ln1_b = (const float*)d_in[9];
    const float* dw_w  = (const float*)d_in[10];
    const float* dw_b  = (const float*)d_in[11];
    const float* ln2_g = (const float*)d_in[12];
    const float* ln2_b = (const float*)d_in[13];
    const float* pw2_w = (const float*)d_in[14];
    const float* pw2_b = (const float*)d_in[15];
    const float* ln3_g = (const float*)d_in[16];
    const float* ln3_b = (const float*)d_in[17];

    const size_t VOL = (size_t)BB * CC * LL;        // 4.19M elements
    bf16_t* qbuf = (bf16_t*)d_ws;                   // all bf16: L2-resident
    bf16_t* Kbuf = qbuf + VOL;
    bf16_t* Vbuf = Kbuf + 4 * VOL;
    bf16_t* t1   = Vbuf + 4 * VOL;
    bf16_t* t2   = t1 + VOL;
    bf16_t* xbuf = t2 + VOL;
    float*  attn = (float*)(xbuf + VOL);            // fp32, 2 MB

    const dim3 gpw(LL / 64, BB);
    const dim3 gdw(LL, BB);

    auto run_block_f32in = [&](int blk, const float* xin, bf16_t* yout,
                               int Ctot, int coff) {
        pw_block_kernel<float, bf16_t, true, false><<<gpw, 256, 0, stream>>>(
            xin, pw1_w + blk * CC * CC, pw1_b + blk * CC,
            ln1_g + blk * CC, ln1_b + blk * CC, nullptr, t1, CC, 0);
        dw_block_kernel<<<gdw, 128, 0, stream>>>(
            t1, dw_w + blk * CC * 27, dw_b + blk * CC,
            ln2_g + blk * CC, ln2_b + blk * CC, t2);
        pw_block_kernel<bf16_t, bf16_t, false, false><<<gpw, 256, 0, stream>>>(
            t2, pw2_w + blk * CC * CC, pw2_b + blk * CC,
            ln3_g + blk * CC, ln3_b + blk * CC, nullptr, yout, Ctot, coff);
    };

    run_block_f32in(0, query, qbuf, CC, 0);
    for (int i = 0; i < 4; ++i) run_block_f32in(1 + i, mods[i], Kbuf, CK, i * CC);
    for (int i = 0; i < 4; ++i) run_block_f32in(5 + i, mods[i], Vbuf, CK, i * CC);

    qk_kernel<<<dim3(CK / 128, BB), 256, 0, stream>>>(qbuf, Kbuf, attn);
    softmax_kernel<<<dim3(BB * CC / 4), 128, 0, stream>>>(attn, mask);
    av_kernel<<<dim3(LL / 128, BB), 256, 0, stream>>>(attn, Vbuf, xbuf);

    // Block 9 (bf16 in) + residual add of fp32 query, fp32 output.
    pw_block_kernel<bf16_t, bf16_t, true, false><<<gpw, 256, 0, stream>>>(
        xbuf, pw1_w + 9 * CC * CC, pw1_b + 9 * CC,
        ln1_g + 9 * CC, ln1_b + 9 * CC, nullptr, t1, CC, 0);
    dw_block_kernel<<<gdw, 128, 0, stream>>>(
        t1, dw_w + 9 * CC * 27, dw_b + 9 * CC,
        ln2_g + 9 * CC, ln2_b + 9 * CC, t2);
    pw_block_kernel<bf16_t, float, false, true><<<gpw, 256, 0, stream>>>(
        t2, pw2_w + 9 * CC * CC, pw2_b + 9 * CC,
        ln3_g + 9 * CC, ln3_b + 9 * CC, query, (float*)d_out, CC, 0);
}
